// SOW_V3_53300544143459
// MI455X (gfx1250) — matrix operand order repaired and checked by reading, not compiled, not run on hardware
//
#include <hip/hip_runtime.h>

typedef __bf16 bf16;
typedef __attribute__((ext_vector_type(16))) __bf16 v16bf;
typedef __attribute__((ext_vector_type(8)))  __bf16 v8bf;
typedef __attribute__((ext_vector_type(8)))  float  v8f;

#define NDIM    512
#define NROUND  511
#define NHALF   256

#define ASTR 40   // LDS row stride in bf16 elems (80B: 16B-aligned, bank-spread)
#define BSTR 40

// ---------------------------------------------------------------------------
// Kernel 1: S = cumprod(sigmoid(a)), one block of 512 threads, LDS scan.
// ---------------------------------------------------------------------------
__global__ __launch_bounds__(512) void sow_scan_sigmoid(const float* __restrict__ a,
                                                        float* __restrict__ S) {
    __shared__ float buf[NDIM];
    int t = threadIdx.x;
    float val = 1.0f / (1.0f + expf(-a[t]));
    buf[t] = val;
    #pragma unroll
    for (int off = 1; off < NDIM; off <<= 1) {
        __syncthreads();
        float prev = (t >= off) ? buf[t - off] : 1.0f;
        __syncthreads();
        val *= prev;
        buf[t] = val;
    }
    S[t] = val;   // S_MAX == 1.0
}

// ---------------------------------------------------------------------------
// Kernel 2: build rotation matrices (rows independent under column-pair
// Givens updates). 8 rows per block in LDS, 511 serial rounds, one disjoint
// pair per thread per round. blockIdx.y: 0 -> V*diag(S), 1 -> Uh.
// ---------------------------------------------------------------------------
__global__ __launch_bounds__(256) void sow_build_rot(const int*   __restrict__ ptbl,
                                                     const float* __restrict__ tan_theta,
                                                     const float* __restrict__ tan_phi,
                                                     const float* __restrict__ S,
                                                     float* __restrict__ Vs,
                                                     float* __restrict__ Uh) {
    __shared__ float rows[8][NDIM];
    const int tid   = threadIdx.x;          // 256
    const int which = blockIdx.y;           // 0 = V, 1 = Uh
    const int base  = blockIdx.x * 8;       // first global row
    const float* tt = which ? tan_phi : tan_theta;

    for (int i = tid; i < 8 * NDIM; i += 256) {
        int r = i >> 9, c = i & (NDIM - 1);
        rows[r][c] = (c == base + r) ? 1.0f : 0.0f;
    }
    __syncthreads();

    for (int r = 0; r < NROUND; ++r) {
        int pi = r * NHALF + tid;
        int p  = ptbl[2 * pi + 0];
        int q  = ptbl[2 * pi + 1];
        float t2  = tt[pi];
        // theta = 0.5*atan(t2): cos(2th)=rsqrt(1+t^2)>0, half-angle identities
        float inv = rsqrtf(1.0f + t2 * t2);
        float c   = sqrtf(0.5f * (1.0f + inv));
        float s   = 0.5f * t2 * inv / c;
        #pragma unroll
        for (int r8 = 0; r8 < 8; ++r8) {
            float x = rows[r8][p];
            float y = rows[r8][q];
            rows[r8][p] = c * x - s * y;
            rows[r8][q] = s * x + c * y;
        }
        __syncthreads();
    }

    float* dst = which ? Uh : Vs;
    for (int i = tid; i < 8 * NDIM; i += 256) {
        int r = i >> 9, c = i & (NDIM - 1);
        float v = rows[r][c];
        if (!which) v *= S[c];               // fold diag(S) into V's columns
        dst[(size_t)(base + r) * NDIM + c] = v;
    }
}

// ---------------------------------------------------------------------------
// Kernel 3: small GEMM  W = (V diag(S)) @ Uh  (512x512x512), f32 staging,
// 3x-bf16-split WMMA. Epilogue writes W pre-split (bf16 hi/lo) AND
// pre-transposed (N-major: WT[n][k]) so the big GEMM's B staging becomes a
// pure contiguous async byte copy.
// ---------------------------------------------------------------------------
__global__ __launch_bounds__(256) void sow_gemm_small(const float* __restrict__ A,
                                                      const float* __restrict__ B,
                                                      bf16* __restrict__ WTH,
                                                      bf16* __restrict__ WTL) {
    __shared__ __align__(16) bf16 Ah[128 * ASTR];
    __shared__ __align__(16) bf16 Al[128 * ASTR];
    __shared__ __align__(16) bf16 Bh[128 * BSTR];
    __shared__ __align__(16) bf16 Bl[128 * BSTR];

    const int tid   = threadIdx.x;
    const int lane  = tid & 31;
    const int wave  = tid >> 5;
    const int wm    = wave >> 2;
    const int wn    = wave & 3;
    const int tileM = blockIdx.y * 128;
    const int tileN = blockIdx.x * 128;

    v8f acc[4][2] = {};

    const int lr  = lane & 15;
    const int kb  = (lane >> 4) * 8;
    const int kbl = (lane >> 4) * 16;

    union Frag { v16bf v; v8bf h[2]; };

    for (int k0 = 0; k0 < NDIM; k0 += 32) {
        #pragma unroll
        for (int i = 0; i < 4; ++i) {
            int idx = tid + i * 256;
            int row = idx >> 3;
            int kk  = (idx & 7) << 2;
            float4 x = *(const float4*)(A + (size_t)(tileM + row) * NDIM + k0 + kk);
            union { bf16 b[4]; uint2 u; } ph, pl;
            ph.b[0] = (bf16)x.x; pl.b[0] = (bf16)(x.x - (float)ph.b[0]);
            ph.b[1] = (bf16)x.y; pl.b[1] = (bf16)(x.y - (float)ph.b[1]);
            ph.b[2] = (bf16)x.z; pl.b[2] = (bf16)(x.z - (float)ph.b[2]);
            ph.b[3] = (bf16)x.w; pl.b[3] = (bf16)(x.w - (float)ph.b[3]);
            *(uint2*)&Ah[row * ASTR + kk] = ph.u;
            *(uint2*)&Al[row * ASTR + kk] = pl.u;
        }
        #pragma unroll
        for (int i = 0; i < 4; ++i) {
            int idx = tid + i * 256;
            int kr  = idx >> 5;
            int nn  = (idx & 31) << 2;
            float4 x = *(const float4*)(B + (size_t)(k0 + kr) * NDIM + tileN + nn);
            bf16 h;
            h = (bf16)x.x; Bh[(nn + 0) * BSTR + kr] = h; Bl[(nn + 0) * BSTR + kr] = (bf16)(x.x - (float)h);
            h = (bf16)x.y; Bh[(nn + 1) * BSTR + kr] = h; Bl[(nn + 1) * BSTR + kr] = (bf16)(x.y - (float)h);
            h = (bf16)x.z; Bh[(nn + 2) * BSTR + kr] = h; Bl[(nn + 2) * BSTR + kr] = (bf16)(x.z - (float)h);
            h = (bf16)x.w; Bh[(nn + 3) * BSTR + kr] = h; Bl[(nn + 3) * BSTR + kr] = (bf16)(x.w - (float)h);
        }
        __syncthreads();

        Frag ah[4], al[4], bh[2], bl[2];
        #pragma unroll
        for (int mt = 0; mt < 4; ++mt) {
            int ro = (wm * 64 + mt * 16 + lr) * ASTR;
            ah[mt].h[0] = *(const v8bf*)&Ah[ro + kb];
            ah[mt].h[1] = *(const v8bf*)&Ah[ro + kb + 16];
            al[mt].h[0] = *(const v8bf*)&Al[ro + kb];
            al[mt].h[1] = *(const v8bf*)&Al[ro + kb + 16];
        }
        #pragma unroll
        for (int nt = 0; nt < 2; ++nt) {
            int co = (wn * 32 + nt * 16 + lr) * BSTR;
            bh[nt].h[0] = *(const v8bf*)&Bh[co + kb];
            bh[nt].h[1] = *(const v8bf*)&Bh[co + kb + 16];
            bl[nt].h[0] = *(const v8bf*)&Bl[co + kb];
            bl[nt].h[1] = *(const v8bf*)&Bl[co + kb + 16];
        }

        #pragma unroll
        for (int mt = 0; mt < 4; ++mt) {
            #pragma unroll
            for (int nt = 0; nt < 2; ++nt) {
                acc[mt][nt] = __builtin_amdgcn_wmma_f32_16x16x32_bf16(
                    false, ah[mt].v, false, bh[nt].v, (short)0, acc[mt][nt], false, false);
                acc[mt][nt] = __builtin_amdgcn_wmma_f32_16x16x32_bf16(
                    false, ah[mt].v, false, bl[nt].v, (short)0, acc[mt][nt], false, false);
                acc[mt][nt] = __builtin_amdgcn_wmma_f32_16x16x32_bf16(
                    false, al[mt].v, false, bh[nt].v, (short)0, acc[mt][nt], false, false);
            }
        }
        __syncthreads();
    }

    // epilogue: split W into bf16 hi/lo, stored N-major: WT[n*512 + k]
    const int col = lane & 15;
    const int rb  = (lane >> 4) * 8;
    #pragma unroll
    for (int mt = 0; mt < 4; ++mt) {
        #pragma unroll
        for (int nt = 0; nt < 2; ++nt) {
            #pragma unroll
            for (int v = 0; v < 8; ++v) {
                int r = tileM + wm * 64 + mt * 16 + rb + v;                // K index
                int c = tileN + wn * 32 + nt * 16 + col;                   // N index
                float  val = acc[mt][nt][v];
                bf16 h = (bf16)val;
                bf16 l = (bf16)(val - (float)h);
                WTH[(size_t)c * NDIM + r] = h;
                WTL[(size_t)c * NDIM + r] = l;
            }
        }
    }
}

// ---------------------------------------------------------------------------
// Kernel 4: big GEMM  out[16384x512] = Z0 @ W.
// A: f32 -> bf16 hi/lo split in-kernel (Z0 lives in L2 across N-tiles).
// B: pre-split, pre-transposed bf16 from WT -- staged with double-buffered
//    GLOBAL_LOAD_ASYNC_TO_LDS_B128 (ASYNCcnt), 4 chunks/wave, each wave owns
//    a disjoint 2KB slice; s_wait_asynccnt + the workgroup barrier publish it.
// ---------------------------------------------------------------------------
__device__ __forceinline__ void sow_async_b128(unsigned lds_addr, unsigned goff,
                                               unsigned long long gbase) {
    asm volatile("global_load_async_to_lds_b128 %0, %1, %2"
                 :: "v"(lds_addr), "v"(goff), "s"(gbase) : "memory");
}

__global__ __launch_bounds__(256) void sow_gemm_big(const float* __restrict__ A,
                                                    const bf16* __restrict__ WTH,
                                                    const bf16* __restrict__ WTL,
                                                    float* __restrict__ C) {
    __shared__ __align__(16) bf16 Ah[128 * ASTR];
    __shared__ __align__(16) bf16 Al[128 * ASTR];
    __shared__ __align__(16) bf16 BsH[2][128 * BSTR];   // double-buffered
    __shared__ __align__(16) bf16 BsL[2][128 * BSTR];

    const int tid   = threadIdx.x;
    const int lane  = tid & 31;
    const int wave  = tid >> 5;
    const int wm    = wave >> 2;
    const int wn    = wave & 3;
    const int tileM = blockIdx.y * 128;
    const int tileN = blockIdx.x * 128;

    v8f acc[4][2] = {};

    const int lr  = lane & 15;
    const int kb  = (lane >> 4) * 8;
    const int kbl = (lane >> 4) * 16;

    union Frag { v16bf v; v8bf h[2]; };

    // --- async B chunk mapping: wave w, chunk i covers (half = w>>2),
    //     row n = (w&3)*32 + i*8 + (lane>>2), kc = lane&3, 16B each.
    const int      halfB = wave >> 2;                        // 0 = hi, 1 = lo
    const unsigned kc    = lane & 3;
    unsigned ldsLocal[4], gOff0[4];
    #pragma unroll
    for (int i = 0; i < 4; ++i) {
        unsigned n  = (unsigned)((wave & 3) * 32 + i * 8 + (lane >> 2));
        ldsLocal[i] = n * (BSTR * 2) + kc * 16;              // bytes in buffer
        gOff0[i]    = n * (NDIM * 2) + kc * 16 + (unsigned)tileN * (NDIM * 2);
    }
    const unsigned bufBase0 = (unsigned)(size_t)(halfB ? &BsL[0][0] : &BsH[0][0]);
    const unsigned bufBase1 = (unsigned)(size_t)(halfB ? &BsL[1][0] : &BsH[1][0]);
    const unsigned long long gbase =
        (unsigned long long)(size_t)(halfB ? WTL : WTH);

    // prologue: issue async copies for K-tile 0 into buffer 0
    #pragma unroll
    for (int i = 0; i < 4; ++i) sow_async_b128(bufBase0 + ldsLocal[i], gOff0[i], gbase);

    for (int kt = 0; kt < NDIM / 32; ++kt) {
        const int k0  = kt * 32;
        const int buf = kt & 1;

        // --- stage A tile 128x32: f32 -> bf16 hi/lo ---
        #pragma unroll
        for (int i = 0; i < 4; ++i) {
            int idx = tid + i * 256;
            int row = idx >> 3;
            int kk  = (idx & 7) << 2;
            float4 x = *(const float4*)(A + (size_t)(tileM + row) * NDIM + k0 + kk);
            union { bf16 b[4]; uint2 u; } ph, pl;
            ph.b[0] = (bf16)x.x; pl.b[0] = (bf16)(x.x - (float)ph.b[0]);
            ph.b[1] = (bf16)x.y; pl.b[1] = (bf16)(x.y - (float)ph.b[1]);
            ph.b[2] = (bf16)x.z; pl.b[2] = (bf16)(x.z - (float)ph.b[2]);
            ph.b[3] = (bf16)x.w; pl.b[3] = (bf16)(x.w - (float)ph.b[3]);
            *(uint2*)&Ah[row * ASTR + kk] = ph.u;
            *(uint2*)&Al[row * ASTR + kk] = pl.u;
        }

        // --- prefetch next B tile into the other buffer, then wait for ours ---
        if (kt + 1 < NDIM / 32) {
            unsigned kadd = (unsigned)(kt + 1) * 64;         // 32 K * 2B
            unsigned nb   = buf ? bufBase0 : bufBase1;
            #pragma unroll
            for (int i = 0; i < 4; ++i) sow_async_b128(nb + ldsLocal[i], gOff0[i] + kadd, gbase);
            asm volatile("s_wait_asynccnt 0x4" ::: "memory"); // current tile done (in-order)
        } else {
            asm volatile("s_wait_asynccnt 0x0" ::: "memory");
        }
        __syncthreads();

        // --- fragments per ISA VGPR layouts ---
        Frag ah[4], al[4], bh[2], bl[2];
        #pragma unroll
        for (int mt = 0; mt < 4; ++mt) {
            int ro = (wm * 64 + mt * 16 + lr) * ASTR;
            ah[mt].h[0] = *(const v8bf*)&Ah[ro + kb];
            ah[mt].h[1] = *(const v8bf*)&Ah[ro + kb + 16];
            al[mt].h[0] = *(const v8bf*)&Al[ro + kb];
            al[mt].h[1] = *(const v8bf*)&Al[ro + kb + 16];
        }
        #pragma unroll
        for (int nt = 0; nt < 2; ++nt) {
            int co = (wn * 32 + nt * 16 + lr) * BSTR;
            bh[nt].h[0] = *(const v8bf*)&BsH[buf][co + kb];
            bh[nt].h[1] = *(const v8bf*)&BsH[buf][co + kb + 16];
            bl[nt].h[0] = *(const v8bf*)&BsL[buf][co + kb];
            bl[nt].h[1] = *(const v8bf*)&BsL[buf][co + kb + 16];
        }

        // --- 3x-split WMMA: hi*hi + hi*lo + lo*hi ---
        #pragma unroll
        for (int mt = 0; mt < 4; ++mt) {
            #pragma unroll
            for (int nt = 0; nt < 2; ++nt) {
                acc[mt][nt] = __builtin_amdgcn_wmma_f32_16x16x32_bf16(
                    false, ah[mt].v, false, bh[nt].v, (short)0, acc[mt][nt], false, false);
                acc[mt][nt] = __builtin_amdgcn_wmma_f32_16x16x32_bf16(
                    false, ah[mt].v, false, bl[nt].v, (short)0, acc[mt][nt], false, false);
                acc[mt][nt] = __builtin_amdgcn_wmma_f32_16x16x32_bf16(
                    false, al[mt].v, false, bh[nt].v, (short)0, acc[mt][nt], false, false);
            }
        }
        __syncthreads();
    }

    // --- epilogue: ISA 16x16 f32 D layout ---
    const int col = lane & 15;
    const int rb  = (lane >> 4) * 8;
    #pragma unroll
    for (int mt = 0; mt < 4; ++mt) {
        #pragma unroll
        for (int nt = 0; nt < 2; ++nt) {
            #pragma unroll
            for (int v = 0; v < 8; ++v) {
                size_t r = (size_t)(tileM + wm * 64 + mt * 16 + rb + v);
                C[r * NDIM + tileN + wn * 32 + nt * 16 + col] = acc[mt][nt][v];
            }
        }
    }
}

// ---------------------------------------------------------------------------
// Launch: out = Z0 @ ((V diag(S)) @ Uh)
// ---------------------------------------------------------------------------
extern "C" void kernel_launch(void* const* d_in, const int* in_sizes, int n_in,
                              void* d_out, int out_size, void* d_ws, size_t ws_size,
                              hipStream_t stream) {
    const float* Z0        = (const float*)d_in[0];  // 16384 x 512
    const float* a         = (const float*)d_in[1];  // 512
    const float* tan_theta = (const float*)d_in[2];  // 511*256
    const float* tan_phi   = (const float*)d_in[3];  // 511*256
    const int*   ptbl      = (const int*)d_in[4];    // 511*256*2
    float*       out       = (float*)d_out;

    char* w = (char*)d_ws;
    float* S   = (float*)(w);                                   // 2 KB
    float* Vp  = (float*)(w + 4096);                            // 1 MB : V*diag(S)
    float* Uh  = (float*)(w + 4096 + (1u << 20));               // 1 MB
    bf16*  WTH = (bf16*) (w + 4096 + (2u << 20));               // 512 KB (N-major)
    bf16*  WTL = (bf16*) (w + 4096 + (2u << 20) + (512u << 10)); // 512 KB

    const int BS = in_sizes[0] / NDIM;                          // 16384 rows

    sow_scan_sigmoid<<<1, 512, 0, stream>>>(a, S);
    sow_build_rot<<<dim3(NDIM / 8, 2), 256, 0, stream>>>(ptbl, tan_theta, tan_phi, S, Vp, Uh);
    sow_gemm_small<<<dim3(NDIM / 128, NDIM / 128), 256, 0, stream>>>(Vp, Uh, WTH, WTL);
    sow_gemm_big<<<dim3(NDIM / 128, BS / 128), 256, 0, stream>>>(Z0, WTH, WTL, out);
}
